// TransformerBlock_13520557047933
// MI455X (gfx1250) — matrix operand order repaired and checked by reading, not compiled, not run on hardware
//
#include <hip/hip_runtime.h>
#include <math.h>

typedef __attribute__((ext_vector_type(16))) __bf16 v16bf;
typedef __attribute__((ext_vector_type(8)))  __bf16 v8bf;
typedef __attribute__((ext_vector_type(8)))  float  v8f;

#define BQ     8
#define LQ     8192
#define CQ     512
#define HQ     8
#define WSQ    32
#define SHIFTQ 16
#define MLPQ   2048
#define NWQ    (LQ / WSQ)     /* 256 windows per batch */
#define MROWS  (BQ * LQ)      /* 65536 token rows      */
#define SAP    40             /* padded LDS row stride in bf16 elems (80 B) */

enum { EP_BF16 = 0, EP_GELU_BF16 = 1, EP_ROLL_RES_F32 = 2, EP_RES_F32 = 3 };

__device__ __forceinline__ v8f wmma_bf16(v16bf a, v16bf b, v8f c) {
  return __builtin_amdgcn_wmma_f32_16x16x32_bf16(false, a, false, b, (short)0, c,
                                                 false, false);
}

// CDNA5 async copy: 16 B per lane, global -> LDS, tracked by ASYNCcnt.
// Per ISA §10.2 the low 32 bits of a flat LDS address are the LDS byte offset.
__device__ __forceinline__ uint32_t lds_u32(const void* p) {
  return (uint32_t)(uintptr_t)p;
}
__device__ __forceinline__ void async_copy_b128(uint32_t lds_off, const void* g) {
  asm volatile("global_load_async_to_lds_b128 %0, %1, off"
               :: "v"(lds_off), "v"(g) : "memory");
}
__device__ __forceinline__ void wait_async0() {
  asm volatile("s_wait_asynccnt 0x0" ::: "memory");
}

// ---------------------------------------------------------------------------
// f32 -> bf16 weight conversion
// ---------------------------------------------------------------------------
__global__ __launch_bounds__(256) void cvt_f32_bf16(const float* __restrict__ in,
                                                    __bf16* __restrict__ out, int n) {
  int i = blockIdx.x * 256 + threadIdx.x;
  if (i < n) out[i] = (__bf16)in[i];
}

// ---------------------------------------------------------------------------
// Fused LayerNorm (+optional roll(-SHIFT)) -> bf16.  One wave32 per row of 512.
// ---------------------------------------------------------------------------
__global__ __launch_bounds__(256) void ln_bf16_kernel(const float* __restrict__ x,
                                                      const float* __restrict__ g,
                                                      const float* __restrict__ be,
                                                      __bf16* __restrict__ out,
                                                      int doRoll) {
  const int row  = blockIdx.x * 8 + (threadIdx.x >> 5);
  const int lane = threadIdx.x & 31;
  int src = row;
  if (doRoll) {
    int b = row >> 13;            // / LQ
    int j = row & (LQ - 1);
    src = b * LQ + ((j + SHIFTQ) & (LQ - 1));
  }
  const float4* p = (const float4*)(x + (size_t)src * CQ) + lane * 4;
  float4 d[4];
  float s = 0.f, sq = 0.f;
#pragma unroll
  for (int i = 0; i < 4; ++i) {
    d[i] = p[i];
    s  += d[i].x + d[i].y + d[i].z + d[i].w;
    sq += d[i].x * d[i].x + d[i].y * d[i].y + d[i].z * d[i].z + d[i].w * d[i].w;
  }
#pragma unroll
  for (int off = 16; off > 0; off >>= 1) {
    s  += __shfl_xor(s, off, 32);
    sq += __shfl_xor(sq, off, 32);
  }
  const float mean = s * (1.0f / CQ);
  const float var  = sq * (1.0f / CQ) - mean * mean;
  const float rs   = rsqrtf(var + 1e-5f);
  __bf16* o = out + (size_t)row * CQ + lane * 16;
  const float* gp = g + lane * 16;
  const float* bp = be + lane * 16;
  float vals[16] = {d[0].x, d[0].y, d[0].z, d[0].w, d[1].x, d[1].y, d[1].z, d[1].w,
                    d[2].x, d[2].y, d[2].z, d[2].w, d[3].x, d[3].y, d[3].z, d[3].w};
#pragma unroll
  for (int e = 0; e < 16; ++e)
    o[e] = (__bf16)((vals[e] - mean) * rs * gp[e] + bp[e]);
}

// ---------------------------------------------------------------------------
// Tiled bf16 WMMA GEMM with double-buffered async-to-LDS staging.
// out[M,N] = A[M,K] @ W[N,K]^T  (+fused epilogues)
// 256 threads = 8 waves; block tile 128x64; 32x32 per wave; K-step 32.
// ---------------------------------------------------------------------------
template <int EP>
__global__ __launch_bounds__(256) void gemm_bf16_wmma(
    const __bf16* __restrict__ A, const __bf16* __restrict__ W,
    const float* __restrict__ bias, const float* __restrict__ resid,
    __bf16* __restrict__ outB, float* __restrict__ outF, int N, int K) {
  __shared__ __bf16 sA[2][128 * SAP];
  __shared__ __bf16 sB[2][64 * SAP];

  const int tid   = threadIdx.x;
  const int lane  = tid & 31;
  const int wave  = tid >> 5;
  const int mBlock = blockIdx.y * 128;
  const int nBlock = blockIdx.x * 64;
  const int l15   = lane & 15;
  const int lhalf = (lane < 16) ? 0 : 1;
  const int ka    = lhalf * 8;    // A-matrix K chunk offset (ISA 16-bit A layout)
  const int kbw   = lhalf * 8;    // B-matrix K chunk offset (ISA 16-bit B layout)

  // producer chunk mapping: 16B chunks, 4 per 32-elem row
  const int arow0 = tid >> 2, acc0 = tid & 3;              // chunk tid
  const int arow1 = (tid + 256) >> 2, acc1 = tid & 3;      // chunk tid+256
  const int brow  = tid >> 2, bcc = tid & 3;               // B has 256 chunks

  uint32_t sA0 = lds_u32(&sA[0][0]), sA1 = lds_u32(&sA[1][0]);
  uint32_t sB0 = lds_u32(&sB[0][0]), sB1 = lds_u32(&sB[1][0]);

  auto issue = [&](int buf, int kb) {
    uint32_t aBase = buf ? sA1 : sA0;
    uint32_t bBase = buf ? sB1 : sB0;
    async_copy_b128(aBase + arow0 * (SAP * 2) + acc0 * 16,
                    A + (size_t)(mBlock + arow0) * K + kb + acc0 * 8);
    async_copy_b128(aBase + arow1 * (SAP * 2) + acc1 * 16,
                    A + (size_t)(mBlock + arow1) * K + kb + acc1 * 8);
    async_copy_b128(bBase + brow * (SAP * 2) + bcc * 16,
                    W + (size_t)(nBlock + brow) * K + kb + bcc * 8);
  };

  v8f acc[2][2] = {};
  const int nk = K >> 5;
  issue(0, 0);

  for (int kt = 0; kt < nk; ++kt) {
    wait_async0();
    __syncthreads();
    if (kt + 1 < nk) issue((kt + 1) & 1, (kt + 1) << 5);

    const __bf16* As = &sA[kt & 1][0];
    const __bf16* Bs = &sB[kt & 1][0];
    const int ar = (wave & 3) * 32 + l15;
    const int br = (wave >> 2) * 32 + l15;

    v16bf at[2], bt[2];
#pragma unroll
    for (int tm = 0; tm < 2; ++tm) {
      const __bf16* ap = As + (ar + tm * 16) * SAP + ka;
      v8bf lo = *(const v8bf*)ap;
      v8bf hi = *(const v8bf*)(ap + 16);
      at[tm] = __builtin_shufflevector(lo, hi, 0, 1, 2, 3, 4, 5, 6, 7, 8, 9, 10, 11,
                                       12, 13, 14, 15);
    }
#pragma unroll
    for (int tn = 0; tn < 2; ++tn) {
      const __bf16* bp = Bs + (br + tn * 16) * SAP + kbw;
      v8bf lo = *(const v8bf*)bp;
      v8bf hi = *(const v8bf*)(bp + 16);
      bt[tn] = __builtin_shufflevector(lo, hi, 0, 1, 2, 3, 4, 5, 6, 7, 8, 9, 10, 11,
                                       12, 13, 14, 15);
    }
    acc[0][0] = wmma_bf16(at[0], bt[0], acc[0][0]);
    acc[0][1] = wmma_bf16(at[0], bt[1], acc[0][1]);
    acc[1][0] = wmma_bf16(at[1], bt[0], acc[1][0]);
    acc[1][1] = wmma_bf16(at[1], bt[1], acc[1][1]);
  }

  const int mBase = mBlock + (wave & 3) * 32;
  const int nBase = nBlock + (wave >> 2) * 32;
#pragma unroll
  for (int tm = 0; tm < 2; ++tm)
#pragma unroll
    for (int tn = 0; tn < 2; ++tn) {
      const int n_g = nBase + tn * 16 + l15;
      float bv = 0.f;
      if constexpr (EP != EP_BF16) bv = bias[n_g];
#pragma unroll
      for (int r = 0; r < 8; ++r) {
        const int m_g = mBase + tm * 16 + lhalf * 8 + r;
        float v = acc[tm][tn][r];
        if constexpr (EP == EP_BF16) {
          outB[(size_t)m_g * N + n_g] = (__bf16)v;
        } else if constexpr (EP == EP_GELU_BF16) {
          float t = v + bv;
          float gl = 0.5f * t * (1.0f + erff(t * 0.70710678118654752f));
          outB[(size_t)m_g * N + n_g] = (__bf16)gl;
        } else if constexpr (EP == EP_ROLL_RES_F32) {
          int b = m_g >> 13;
          int j = m_g & (LQ - 1);
          int i = (j + SHIFTQ) & (LQ - 1);
          size_t idx = ((size_t)(b * LQ + i)) * N + n_g;
          outF[idx] = resid[idx] + v + bv;
        } else {  // EP_RES_F32
          size_t idx = (size_t)m_g * N + n_g;
          outF[idx] = resid[idx] + v + bv;
        }
      }
    }
}

// ---------------------------------------------------------------------------
// Windowed attention: one wave per (window, head).
// v tile is async-DMA'd to LDS at entry and consumed after the softmax,
// hiding the copy behind the q@k^T WMMAs and the double softmax.
// ---------------------------------------------------------------------------
__global__ __launch_bounds__(256) void attn_wmma_kernel(
    const __bf16* __restrict__ qkv, const float* __restrict__ rel_table,
    __bf16* __restrict__ obf) {
  __shared__ float  smem[8][32][33];
  __shared__ __bf16 svt[8][32 * 64];
  const int lane  = threadIdx.x & 31;
  const int wave  = threadIdx.x >> 5;
  const int task  = blockIdx.x * 8 + wave;
  const int wg    = task >> 3;       // window index within B*nW
  const int h     = task & 7;        // head
  const int wi    = wg & (NWQ - 1);  // window index within a batch
  const int rowBase = wg * WSQ;
  const int l15   = lane & 15;
  const int lhalf = (lane < 16) ? 0 : 1;
  const int QKVS  = 3 * CQ;  // row stride of packed qkv

  const __bf16* qp = qkv + (size_t)rowBase * QKVS + h * 64;
  const __bf16* kp = qp + CQ;
  const __bf16* vp = qp + 2 * CQ;

  // ---- kick off async v-tile copy (32 x 64 bf16 -> LDS) ----
  {
    const uint32_t vb = lds_u32(&svt[wave][0]);
#pragma unroll
    for (int c = 0; c < 8; ++c) {
      int chunk = c * 32 + lane;   // 256 x 16B chunks
      int row = chunk >> 3;
      int cc  = chunk & 7;
      async_copy_b128(vb + row * 128 + cc * 16,
                      vp + (size_t)row * QKVS + cc * 8);
    }
  }

  // ---- S = q @ k^T  (2x2 tiles, K=64 in two steps) ----
  v8f s[2][2] = {};
  for (int kk = 0; kk < 64; kk += 32) {
    v16bf aq[2], bk[2];
#pragma unroll
    for (int tm = 0; tm < 2; ++tm) {
      const __bf16* ar = qp + (size_t)(tm * 16 + l15) * QKVS + kk + lhalf * 16;
      v8bf lo = *(const v8bf*)ar;
      v8bf hi = *(const v8bf*)(ar + 8);
      aq[tm] = __builtin_shufflevector(lo, hi, 0, 1, 2, 3, 4, 5, 6, 7, 8, 9, 10, 11,
                                       12, 13, 14, 15);
    }
#pragma unroll
    for (int tn = 0; tn < 2; ++tn)
      bk[tn] = *(const v16bf*)(kp + (size_t)(tn * 16 + l15) * QKVS + kk + lhalf * 16);
#pragma unroll
    for (int tm = 0; tm < 2; ++tm)
#pragma unroll
      for (int tn = 0; tn < 2; ++tn) s[tm][tn] = wmma_bf16(aq[tm], bk[tn], s[tm][tn]);
  }

  // ---- stash scaled scores to LDS ----
  const float scale = 0.125f;  // 64^-0.5
#pragma unroll
  for (int tm = 0; tm < 2; ++tm)
#pragma unroll
    for (int tn = 0; tn < 2; ++tn)
#pragma unroll
      for (int r = 0; r < 8; ++r)
        smem[wave][tm * 16 + lhalf * 8 + r][tn * 16 + l15] = s[tm][tn][r] * scale;
  __syncthreads();

  // ---- faithful double softmax, row m = lane ----
  {
    const int m = lane;
    float* row = smem[wave][m];
    float mx = row[0];
#pragma unroll
    for (int n = 1; n < 32; ++n) mx = fmaxf(mx, row[n]);
    float t[32];
    float sum = 0.f;
#pragma unroll
    for (int n = 0; n < 32; ++n) {
      t[n] = __expf(row[n] - mx);
      sum += t[n];
    }
    const float inv = 1.0f / sum;
    const bool maskedWin = (wi == NWQ - 1);
    float mx2 = -1e30f;
#pragma unroll
    for (int n = 0; n < 32; ++n) {
      float a = t[n] * inv + rel_table[(m - n + WSQ - 1) * HQ + h];
      if (maskedWin && ((m < 16) != (n < 16))) a -= 100.0f;
      t[n] = a;
      mx2 = fmaxf(mx2, a);
    }
    float sum2 = 0.f;
#pragma unroll
    for (int n = 0; n < 32; ++n) {
      t[n] = __expf(t[n] - mx2);
      sum2 += t[n];
    }
    const float inv2 = 1.0f / sum2;
#pragma unroll
    for (int n = 0; n < 32; ++n) row[n] = t[n] * inv2;
  }
  __syncthreads();

  // ---- O = att @ v  (2x4 tiles, K=32 single step); v comes from LDS ----
  wait_async0();  // v tile resident (per-wave produce/consume)
  v16bf at[2];
#pragma unroll
  for (int tm = 0; tm < 2; ++tm) {
    const int m = tm * 16 + l15;
    const int k0 = lhalf * 16;
    v16bf a;
#pragma unroll
    for (int e = 0; e < 8; ++e) {
      a[e]     = (__bf16)smem[wave][m][k0 + e];
      a[8 + e] = (__bf16)smem[wave][m][k0 + 8 + e];
    }
    at[tm] = a;
  }
  v8f o[2][4] = {};
#pragma unroll
  for (int tn = 0; tn < 4; ++tn) {
    const int dd  = tn * 16 + l15;
    const int kr0 = lhalf * 16;
    v16bf bt;
#pragma unroll
    for (int kk2 = 0; kk2 < 16; ++kk2)
      bt[kk2] = svt[wave][(kr0 + kk2) * 64 + dd];
#pragma unroll
    for (int tm = 0; tm < 2; ++tm) o[tm][tn] = wmma_bf16(at[tm], bt, o[tm][tn]);
  }
#pragma unroll
  for (int tm = 0; tm < 2; ++tm)
#pragma unroll
    for (int tn = 0; tn < 4; ++tn)
#pragma unroll
      for (int r = 0; r < 8; ++r) {
        const int m = tm * 16 + lhalf * 8 + r;
        const int n = tn * 16 + l15;
        obf[(size_t)(rowBase + m) * CQ + h * 64 + n] = (__bf16)o[tm][tn][r];
      }
}

// ---------------------------------------------------------------------------
extern "C" void kernel_launch(void* const* d_in, const int* in_sizes, int n_in,
                              void* d_out, int out_size, void* d_ws, size_t ws_size,
                              hipStream_t stream) {
  (void)in_sizes; (void)n_in; (void)out_size; (void)ws_size;
  const float* x         = (const float*)d_in[0];
  const float* w_qkv     = (const float*)d_in[1];
  const float* w_out     = (const float*)d_in[2];
  const float* b_out     = (const float*)d_in[3];
  const float* rel_table = (const float*)d_in[4];
  const float* g1        = (const float*)d_in[5];
  const float* be1       = (const float*)d_in[6];
  const float* g2        = (const float*)d_in[7];
  const float* be2       = (const float*)d_in[8];
  const float* w_fc1     = (const float*)d_in[9];
  const float* b_fc1     = (const float*)d_in[10];
  const float* w_fc2     = (const float*)d_in[11];
  const float* b_fc2     = (const float*)d_in[12];
  float* out = (float*)d_out;

  size_t off = 0;
  auto alloc = [&](size_t bytes) -> void* {
    void* r = (char*)d_ws + off;
    off += (bytes + 255) & ~(size_t)255;
    return r;
  };
  __bf16* wqkv_bf = (__bf16*)alloc((size_t)3 * CQ * CQ * 2);
  __bf16* wout_bf = (__bf16*)alloc((size_t)CQ * CQ * 2);
  __bf16* wfc1_bf = (__bf16*)alloc((size_t)MLPQ * CQ * 2);
  __bf16* wfc2_bf = (__bf16*)alloc((size_t)CQ * MLPQ * 2);
  __bf16* buf0    = (__bf16*)alloc((size_t)MROWS * CQ * 2);      // h_bf, then o_bf
  __bf16* qkv_bf  = (__bf16*)alloc((size_t)MROWS * 3 * CQ * 2);  // qkv, then h2_bf
  float*  y       = (float*)alloc((size_t)MROWS * CQ * 4);
  __bf16* m_bf    = (__bf16*)alloc((size_t)MROWS * MLPQ * 2);

  cvt_f32_bf16<<<(3 * CQ * CQ + 255) / 256, 256, 0, stream>>>(w_qkv, wqkv_bf, 3 * CQ * CQ);
  cvt_f32_bf16<<<(CQ * CQ + 255) / 256, 256, 0, stream>>>(w_out, wout_bf, CQ * CQ);
  cvt_f32_bf16<<<(MLPQ * CQ + 255) / 256, 256, 0, stream>>>(w_fc1, wfc1_bf, MLPQ * CQ);
  cvt_f32_bf16<<<(CQ * MLPQ + 255) / 256, 256, 0, stream>>>(w_fc2, wfc2_bf, CQ * MLPQ);

  // h = bf16(roll(LN1(x), -SHIFT))
  ln_bf16_kernel<<<MROWS / 8, 256, 0, stream>>>(x, g1, be1, buf0, 1);

  // qkv = h @ w_qkv^T     (M=65536, N=1536, K=512)
  gemm_bf16_wmma<EP_BF16><<<dim3((3 * CQ) / 64, MROWS / 128), 256, 0, stream>>>(
      buf0, wqkv_bf, nullptr, nullptr, qkv_bf, nullptr, 3 * CQ, CQ);

  // windowed attention -> o (bf16, reuses buf0)
  attn_wmma_kernel<<<(MROWS / WSQ) * HQ / 8, 256, 0, stream>>>(qkv_bf, rel_table, buf0);

  // y = x + roll(o @ w_out^T + b_out, +SHIFT)
  gemm_bf16_wmma<EP_ROLL_RES_F32><<<dim3(CQ / 64, MROWS / 128), 256, 0, stream>>>(
      buf0, wout_bf, b_out, x, nullptr, y, CQ, CQ);

  // h2 = bf16(LN2(y))   (reuses qkv buffer)
  ln_bf16_kernel<<<MROWS / 8, 256, 0, stream>>>(y, g2, be2, qkv_bf, 0);

  // m = bf16(gelu(h2 @ w_fc1^T + b_fc1))   (M=65536, N=2048, K=512)
  gemm_bf16_wmma<EP_GELU_BF16><<<dim3(MLPQ / 64, MROWS / 128), 256, 0, stream>>>(
      qkv_bf, wfc1_bf, b_fc1, nullptr, m_bf, nullptr, MLPQ, CQ);

  // out = y + m @ w_fc2^T + b_fc2   (M=65536, N=512, K=2048)
  gemm_bf16_wmma<EP_RES_F32><<<dim3(CQ / 64, MROWS / 128), 256, 0, stream>>>(
      m_bf, wfc2_bf, b_fc2, y, nullptr, out, CQ, MLPQ);
}
